// Attention_81277961109785
// MI455X (gfx1250) — matrix operand order repaired and checked by reading, not compiled, not run on hardware
//
#include <hip/hip_runtime.h>

// ---------------------------------------------------------------------------
// Channel-attention (Restormer/XCiT style) for MI455X (gfx1250, wave32, WMMA)
//   b=8, c=192, h=w=128 (hw=16384), heads=8, cd=24, 3c=576
// Pipeline:
//   1) qkv = w_qkv @ x            : bf16 WMMA GEMM  (out bf16)
//   2) depthwise 3x3 conv         : VALU            (out bf16)
//   3) row L2 norms of q,k        : reduction
//   4) fused channel attention    : WMMA (qk^T), softmax, WMMA (P v)
//   5) out = w_proj @ attn_out    : bf16 WMMA GEMM  (out fp32)
// ---------------------------------------------------------------------------

typedef __attribute__((ext_vector_type(16))) __bf16 v16bf;
typedef __attribute__((ext_vector_type(8)))  float  v8f;

union Frag16 { v16bf v; uint4 q[2]; };          // 16 bf16 = 32B WMMA operand
union Pack8  { unsigned short s[8];  uint4 q;    };
union Pack32 { unsigned short s[32]; uint4 q[4]; };

__device__ __forceinline__ unsigned short f2bf(float f) {
    unsigned u = __builtin_bit_cast(unsigned, f);
    u += 0x7fffu + ((u >> 16) & 1u);            // round-to-nearest-even
    return (unsigned short)(u >> 16);
}
__device__ __forceinline__ float bf2f(unsigned short s) {
    return __builtin_bit_cast(float, ((unsigned)s) << 16);
}
__device__ __forceinline__ uint4 zero4() { uint4 z; z.x = z.y = z.z = z.w = 0u; return z; }

// ---------------------------------------------------------------------------
// Generic batched GEMM:  C[bi] = A (shared) * B[bi]
//   A: fp32 [M x K] row-major (weights)
//   B: [K x N] row-major, fp32 or bf16 (template)
//   C: [M x N], bf16 or fp32 (template)
// Block tile 64(M) x 256(N), K-step 32, 8 waves of 32x64 each.
// ---------------------------------------------------------------------------
template <bool B_BF16, bool OUT_BF16>
__global__ __launch_bounds__(256) void gemm_wmma(const float* __restrict__ A,
                                                 const void*  __restrict__ Bp,
                                                 void*        __restrict__ Cp,
                                                 int M, int K, int N) {
    __shared__ unsigned short sA[64][40];    // row-major, +8 pad (bank spread)
    __shared__ unsigned short sBt[256][40];  // transposed [n][k], +8 pad

    const int tid  = threadIdx.x;
    const int lane = tid & 31;
    const int wid  = tid >> 5;
    const int hi   = lane >> 4;      // 0: lanes 0-15, 1: lanes 16-31
    const int lrow = lane & 15;
    const int n0   = blockIdx.x * 256;
    const int m0   = blockIdx.y * 64;
    const int bi   = blockIdx.z;
    const int wm   = (wid >> 2) * 32;
    const int wn   = (wid & 3) * 64;

    const float*          Bf = (const float*)Bp          + (size_t)bi * K * N;
    const unsigned short* Bh = (const unsigned short*)Bp + (size_t)bi * K * N;

    v8f acc[2][4];
#pragma unroll
    for (int i = 0; i < 2; ++i)
#pragma unroll
        for (int j = 0; j < 4; ++j)
#pragma unroll
            for (int r = 0; r < 8; ++r) acc[i][j][r] = 0.0f;

    for (int kk = 0; kk < K; kk += 32) {
        __syncthreads();
        // ---- stage A tile [64 x 32] (fp32 -> bf16) -------------------------
        {
            const int row = tid >> 2;
            const int g   = (tid & 3) * 8;
            const float* ap = A + (size_t)(m0 + row) * K + kk + g;
            Pack8 t;
#pragma unroll
            for (int e = 0; e < 8; ++e) t.s[e] = f2bf(ap[e]);
            *(uint4*)&sA[row][g] = t.q;
        }
        // ---- stage B tile [32 x 256] transposed into sBt[n][k] -------------
        {
            Pack32 col;
#pragma unroll
            for (int k2 = 0; k2 < 32; ++k2) {
                const size_t gi = (size_t)(kk + k2) * N + n0 + tid;
                if constexpr (B_BF16) col.s[k2] = Bh[gi];
                else                  col.s[k2] = f2bf(Bf[gi]);
            }
#pragma unroll
            for (int j = 0; j < 4; ++j) *(uint4*)&sBt[tid][j * 8] = col.q[j];
        }
        if (kk + 32 < K) {   // prefetch next B slab (global_prefetch_b8)
            if constexpr (B_BF16) __builtin_prefetch(&Bh[(size_t)(kk + 32) * N + n0 + tid]);
            else                  __builtin_prefetch(&Bf[(size_t)(kk + 32) * N + n0 + tid]);
        }
        __syncthreads();

        // ---- WMMA fragments + 8 MACs per wave ------------------------------
        Frag16 a[2], b[4];
#pragma unroll
        for (int i = 0; i < 2; ++i) {
            const unsigned short* p = &sA[wm + i * 16 + lrow][hi * 8];
            a[i].q[0] = *(const uint4*)p;
            a[i].q[1] = *(const uint4*)(p + 16);
        }
#pragma unroll
        for (int j = 0; j < 4; ++j) {
            const unsigned short* p = &sBt[wn + j * 16 + lrow][hi * 8];
            b[j].q[0] = *(const uint4*)p;
            b[j].q[1] = *(const uint4*)(p + 16);
        }
#pragma unroll
        for (int i = 0; i < 2; ++i)
#pragma unroll
            for (int j = 0; j < 4; ++j)
                acc[i][j] = __builtin_amdgcn_wmma_f32_16x16x32_bf16(
                    false, a[i].v, false, b[j].v, (short)0, acc[i][j], false, false);
    }

    // ---- epilogue ----------------------------------------------------------
    float*          Cf = (float*)Cp          + (size_t)bi * M * N;
    unsigned short* Ch = (unsigned short*)Cp + (size_t)bi * M * N;
#pragma unroll
    for (int i = 0; i < 2; ++i)
#pragma unroll
        for (int j = 0; j < 4; ++j)
#pragma unroll
            for (int r = 0; r < 8; ++r) {
                const int m = m0 + wm + i * 16 + hi * 8 + r;
                const int n = n0 + wn + j * 16 + lrow;
                const size_t off = (size_t)m * N + n;
                if constexpr (OUT_BF16) Ch[off] = f2bf(acc[i][j][r]);
                else                    Cf[off] = acc[i][j][r];
            }
}

// ---------------------------------------------------------------------------
// Depthwise 3x3, SAME padding. One block = 256 contiguous pixels of one chan.
// ---------------------------------------------------------------------------
__global__ __launch_bounds__(256) void dwconv3x3(const unsigned short* __restrict__ in,
                                                 const float* __restrict__ wdw,
                                                 unsigned short* __restrict__ out) {
    const int chan_lin = blockIdx.x >> 6;          // 16384/256 = 64 blocks/chan
    const int ch = chan_lin % 576;
    const int b  = chan_lin / 576;
    const int px = ((blockIdx.x & 63) << 8) + threadIdx.x;
    const int y = px >> 7, x = px & 127;

    const float* wp = wdw + ch * 9;                // uniform -> scalar loads
    const unsigned short* ip = in + ((size_t)b * 576 + ch) * 16384;

    float s = 0.0f;
#pragma unroll
    for (int dy = -1; dy <= 1; ++dy)
#pragma unroll
        for (int dx = -1; dx <= 1; ++dx) {
            const int yy = y + dy, xx = x + dx;
            if (yy >= 0 && yy < 128 && xx >= 0 && xx < 128)
                s += wp[(dy + 1) * 3 + (dx + 1)] * bf2f(ip[yy * 128 + xx]);
        }
    out[((size_t)b * 576 + ch) * 16384 + px] = f2bf(s);
}

// ---------------------------------------------------------------------------
// inv L2 norm of q,k rows: invn[b*384 + r] over qkvd row (b*576 + r), r<384
// ---------------------------------------------------------------------------
__global__ __launch_bounds__(256) void rownorm(const unsigned short* __restrict__ qkvd,
                                               float* __restrict__ invn) {
    const int row = blockIdx.x;                    // 0 .. 8*384-1
    const int bi = row / 384, r = row % 384;
    const unsigned short* p = qkvd + ((size_t)bi * 576 + r) * 16384;
    float s = 0.0f;
    for (int i = threadIdx.x; i < 16384; i += 256) {
        const float v = bf2f(p[i]);
        s += v * v;
    }
    __shared__ float red[256];
    red[threadIdx.x] = s;
    __syncthreads();
    for (int off = 128; off > 0; off >>= 1) {
        if (threadIdx.x < off) red[threadIdx.x] += red[threadIdx.x + off];
        __syncthreads();
    }
    if (threadIdx.x == 0) invn[row] = 1.0f / fmaxf(sqrtf(red[0]), 1e-12f);
}

// ---------------------------------------------------------------------------
// Fused channel attention: one block per (batch, head). cd=24 padded to 32.
// Phase 1: S = q k^T over K=16384 (WMMA, per-wave partials -> LDS reduce)
//          scale by invq*invk*temperature, softmax over 24 cols -> P (bf16 LDS)
// Phase 2: out = P v (WMMA, K=32 with zero padding), bf16 out
// ---------------------------------------------------------------------------
__global__ __launch_bounds__(256) void attn_kernel(const unsigned short* __restrict__ qkvd,
                                                   const float* __restrict__ invn,
                                                   const float* __restrict__ temperature,
                                                   unsigned short* __restrict__ oattn) {
    __shared__ float          sS[8][32][32];   // per-wave partial logits
    __shared__ unsigned short sP[32][40];      // softmax(P) bf16, padded rows
    __shared__ unsigned short sVt[256][40];    // v tile transposed [n][d]

    const int tid = threadIdx.x, lane = tid & 31, wid = tid >> 5;
    const int hi = lane >> 4, lrow = lane & 15;
    const int head = blockIdx.x & 7, bi = blockIdx.x >> 3;
    const size_t RS = 16384;

    const unsigned short* qb = qkvd + ((size_t)bi * 576 +       head * 24) * RS;
    const unsigned short* kb = qkvd + ((size_t)bi * 576 + 192 + head * 24) * RS;
    const unsigned short* vb = qkvd + ((size_t)bi * 576 + 384 + head * 24) * RS;

    // ---- phase 1: q k^T --------------------------------------------------
    v8f acc[2][2];
#pragma unroll
    for (int i = 0; i < 2; ++i)
#pragma unroll
        for (int j = 0; j < 2; ++j)
#pragma unroll
            for (int r = 0; r < 8; ++r) acc[i][j][r] = 0.0f;

    for (int kc = wid; kc < 512; kc += 8) {      // 512 chunks of K=32 over hw
        const int hw0 = kc * 32;
        Frag16 a[2], b2[2];
#pragma unroll
        for (int i = 0; i < 2; ++i) {
            const int mch = i * 16 + lrow;
            if (mch < 24) {
                const unsigned short* p = qb + (size_t)mch * RS + hw0 + hi * 8;
                a[i].q[0] = *(const uint4*)p;
                a[i].q[1] = *(const uint4*)(p + 16);
            } else { a[i].q[0] = zero4(); a[i].q[1] = zero4(); }
        }
#pragma unroll
        for (int j = 0; j < 2; ++j) {
            const int nch = j * 16 + lrow;
            if (nch < 24) {
                const unsigned short* p = kb + (size_t)nch * RS + hw0 + hi * 8;
                b2[j].q[0] = *(const uint4*)p;
                b2[j].q[1] = *(const uint4*)(p + 16);
            } else { b2[j].q[0] = zero4(); b2[j].q[1] = zero4(); }
        }
#pragma unroll
        for (int i = 0; i < 2; ++i)
#pragma unroll
            for (int j = 0; j < 2; ++j)
                acc[i][j] = __builtin_amdgcn_wmma_f32_16x16x32_bf16(
                    false, a[i].v, false, b2[j].v, (short)0, acc[i][j], false, false);
    }
#pragma unroll
    for (int i = 0; i < 2; ++i)
#pragma unroll
        for (int j = 0; j < 2; ++j)
#pragma unroll
            for (int r = 0; r < 8; ++r)
                sS[wid][i * 16 + hi * 8 + r][j * 16 + lrow] = acc[i][j][r];
    __syncthreads();

    // ---- reduce partials, scale, softmax (one thread per row) -------------
    if (tid < 32) {
        const int m = tid;
        if (m < 24) {
            const float iq = invn[(size_t)bi * 384 + head * 24 + m];
            const float tp = temperature[head];
            float s[24], mx = -3.0e38f;
#pragma unroll
            for (int n = 0; n < 24; ++n) {
                float a2 = 0.0f;
#pragma unroll
                for (int w = 0; w < 8; ++w) a2 += sS[w][m][n];
                a2 *= iq * invn[(size_t)bi * 384 + 192 + head * 24 + n] * tp;
                s[n] = a2;
                mx = fmaxf(mx, a2);
            }
            float sum = 0.0f;
#pragma unroll
            for (int n = 0; n < 24; ++n) { s[n] = __expf(s[n] - mx); sum += s[n]; }
            const float inv = 1.0f / sum;
#pragma unroll
            for (int n = 0; n < 24; ++n) sP[m][n] = f2bf(s[n] * inv);
#pragma unroll
            for (int n = 24; n < 32; ++n) sP[m][n] = 0;
        } else {
#pragma unroll
            for (int n = 0; n < 32; ++n) sP[m][n] = 0;
        }
    }
    __syncthreads();

    // ---- phase 2: out = P v ----------------------------------------------
    Frag16 pa[2];
#pragma unroll
    for (int i = 0; i < 2; ++i) {
        const unsigned short* p = &sP[i * 16 + lrow][hi * 8];
        pa[i].q[0] = *(const uint4*)p;
        pa[i].q[1] = *(const uint4*)(p + 16);
    }
    const int nw = wid * 32;

    for (int n0 = 0; n0 < 16384; n0 += 256) {
        __syncthreads();
        {   // stage v tile transposed (channels 24..31 zero-padded)
            Pack32 col;
#pragma unroll
            for (int d = 0; d < 32; ++d)
                col.s[d] = (d < 24) ? vb[(size_t)d * RS + n0 + tid] : (unsigned short)0;
#pragma unroll
            for (int j = 0; j < 4; ++j) *(uint4*)&sVt[tid][j * 8] = col.q[j];
        }
        __syncthreads();

        Frag16 vf[2];
#pragma unroll
        for (int j = 0; j < 2; ++j) {
            const unsigned short* p = &sVt[nw + j * 16 + lrow][hi * 8];
            vf[j].q[0] = *(const uint4*)p;
            vf[j].q[1] = *(const uint4*)(p + 16);
        }
        v8f oa[2][2];
#pragma unroll
        for (int i = 0; i < 2; ++i)
#pragma unroll
            for (int j = 0; j < 2; ++j) {
#pragma unroll
                for (int r = 0; r < 8; ++r) oa[i][j][r] = 0.0f;
                oa[i][j] = __builtin_amdgcn_wmma_f32_16x16x32_bf16(
                    false, pa[i].v, false, vf[j].v, (short)0, oa[i][j], false, false);
            }
#pragma unroll
        for (int i = 0; i < 2; ++i)
#pragma unroll
            for (int j = 0; j < 2; ++j)
#pragma unroll
                for (int r = 0; r < 8; ++r) {
                    const int m = i * 16 + hi * 8 + r;
                    if (m < 24) {
                        const size_t o = ((size_t)bi * 192 + head * 24 + m) * RS +
                                         n0 + nw + j * 16 + lrow;
                        oattn[o] = f2bf(oa[i][j][r]);
                    }
                }
    }
}

// ---------------------------------------------------------------------------
extern "C" void kernel_launch(void* const* d_in, const int* in_sizes, int n_in,
                              void* d_out, int out_size, void* d_ws, size_t ws_size,
                              hipStream_t stream) {
    const float* x      = (const float*)d_in[0];
    const float* w_qkv  = (const float*)d_in[1];
    const float* w_dw   = (const float*)d_in[2];
    const float* w_proj = (const float*)d_in[3];
    const float* temp   = (const float*)d_in[4];
    float* out = (float*)d_out;

    const size_t QKV_BYTES = (size_t)8 * 576 * 16384 * 2;   // 150,994,944
    char* ws = (char*)d_ws;
    unsigned short* qkv_bf  = (unsigned short*)ws;
    unsigned short* qkvd_bf = (unsigned short*)(ws + QKV_BYTES);
    float*          invn    = (float*)(ws + 2 * QKV_BYTES);
    unsigned short* oattn   = qkv_bf;   // qkv_bf is dead after dwconv -> alias

    // 1) qkv 1x1 conv: [576x192] x [192x16384] per batch, out bf16
    gemm_wmma<false, true><<<dim3(64, 9, 8), 256, 0, stream>>>(
        w_qkv, (const void*)x, (void*)qkv_bf, 576, 192, 16384);
    // 2) depthwise 3x3
    dwconv3x3<<<dim3(8 * 576 * 64), 256, 0, stream>>>(qkv_bf, w_dw, qkvd_bf);
    // 3) inverse L2 norms of q,k rows
    rownorm<<<dim3(8 * 384), 256, 0, stream>>>(qkvd_bf, invn);
    // 4) fused channel attention (one block per batch*head)
    attn_kernel<<<dim3(64), 256, 0, stream>>>(qkvd_bf, invn, temp, oattn);
    // 5) proj 1x1 conv: [192x192] x [192x16384] per batch, out fp32
    gemm_wmma<true, false><<<dim3(64, 3, 8), 256, 0, stream>>>(
        w_proj, (const void*)oattn, (void*)out, 192, 192, 16384);
}
